// MDTA_21912923144224
// MI455X (gfx1250) — matrix operand order repaired and checked by reading, not compiled, not run on hardware
//
#include <hip/hip_runtime.h>
#include <math.h>

// ---------------- problem constants ----------------
#define BB   4
#define CC   192
#define C3   576           // 3*C
#define HH   256
#define WW   256
#define HWP  65536         // H*W
#define NH   4             // heads
#define DD   48            // C / heads

// ---------------- CDNA5 vector types ----------------
typedef __attribute__((ext_vector_type(16))) __bf16 v16bf;
typedef __attribute__((ext_vector_type(8)))  float  v8f;
typedef __attribute__((ext_vector_type(8)))  unsigned short u16x8;

union BF16x16 { v16bf v; unsigned short u[16]; u16x8 h[2]; };

__device__ __forceinline__ unsigned short f2bf(float f) {
    union { float f; unsigned u; } x; x.f = f;
    unsigned r = x.u + 0x7FFFu + ((x.u >> 16) & 1u);   // round-to-nearest-even
    return (unsigned short)(r >> 16);
}
__device__ __forceinline__ float bf2f(unsigned short s) {
    union { unsigned u; float f; } x; x.u = ((unsigned)s) << 16;
    return x.f;
}
__device__ __forceinline__ v8f v8f_zero() {
    v8f z = {0.f,0.f,0.f,0.f,0.f,0.f,0.f,0.f};
    return z;
}

// =====================================================================
// Kernel: zero a float region (sumsq + attn accumulators each launch)
// =====================================================================
__global__ void zero_f32_kernel(float* __restrict__ p, int n) {
    int i = blockIdx.x * 256 + threadIdx.x;
    if (i < n) p[i] = 0.f;
}

// =====================================================================
// Generic WMMA GEMM:  Out[b] = A[b] (MxK, f32 row-major) * Bsrc[b] (KxN)
//   BF32  : B source is fp32 (else bf16-as-u16)
//   OUTBF : output is bf16-as-u16 (else fp32)
// Block = 256 thr (8 waves, 2x4), tile = 64(M) x 128(N), K step 32.
// B tile is staged to LDS transposed ([p][c]) with fp32->bf16 convert so
// v_wmma B fragments are contiguous 32B ds loads.  DOUBLE-BUFFERED: the
// next tile's global loads are issued before the current WMMA block so
// their latency is hidden by matrix math; one barrier per K-step.
// =====================================================================
template <bool BF32, bool OUTBF>
__global__ __launch_bounds__(256) void gemm_wmma_kernel(
    const float* __restrict__ A, const void* __restrict__ Bsrc,
    void* __restrict__ Out, int M, int K, long N,
    long aBatch, long bBatch, long oBatch)
{
    __shared__ alignas(32) unsigned short ldsB[2][128 * 32];  // 2 x 8 KB [p][c]

    const int b     = blockIdx.z;
    const float* Ab = A + (size_t)b * aBatch;
    const int tid   = threadIdx.x;
    const int lane  = tid & 31;
    const int wave  = tid >> 5;
    const int wm    = wave >> 2;          // 0..1  (M)
    const int wn    = wave & 3;           // 0..3  (N)
    const int  Mbase = blockIdx.y * 64;
    const long Nbase = (long)blockIdx.x * 128;

    // staging thread mapping: K-pair c, N-subtile p0
    const int sc  = (tid >> 4) * 2;       // 0..30
    const int sp0 = (tid & 15) * 8;       // 0..120

    v8f acc[2][2];
    acc[0][0] = v8f_zero(); acc[0][1] = v8f_zero();
    acc[1][0] = v8f_zero(); acc[1][1] = v8f_zero();

    const int aRow = Mbase + wm * 32 + (lane & 15);
    const int aOff = (lane >> 4) * 16;    // K sub-offset for A fragment
    const int pl   = wn * 32 + (lane & 15);
    const int kh   = (lane >> 4) * 16;    // K half for B fragment

    // ---- prologue: stage tile k0=0 into buffer 0 ----
    {
        const size_t ro = (size_t)b * bBatch + (size_t)sc * N + Nbase + sp0;
        if (BF32) {
            const float* r0 = (const float*)Bsrc + ro;
            const float* r1 = r0 + N;
#pragma unroll
            for (int j = 0; j < 8; ++j) {
                unsigned lo = f2bf(r0[j]), hi = f2bf(r1[j]);
                *(unsigned*)&ldsB[0][(sp0 + j) * 32 + sc] = lo | (hi << 16);
            }
        } else {
            const unsigned short* r0 = (const unsigned short*)Bsrc + ro;
            const unsigned short* r1 = r0 + N;
#pragma unroll
            for (int j = 0; j < 8; ++j)
                *(unsigned*)&ldsB[0][(sp0 + j) * 32 + sc] =
                    (unsigned)r0[j] | ((unsigned)r1[j] << 16);
        }
    }
    __syncthreads();

    int buf = 0;
    for (int k0 = 0; k0 < K; k0 += 32) {
        const bool hasNext = (k0 + 32) < K;

        // ---- B fragments from current LDS buffer (contiguous 32B) ----
        BF16x16 bfr[2];
#pragma unroll
        for (int tj = 0; tj < 2; ++tj) {
            const unsigned short* bp = &ldsB[buf][(pl + tj * 16) * 32 + kh];
            bfr[tj].h[0] = *(const u16x8*)bp;
            bfr[tj].h[1] = *(const u16x8*)(bp + 8);
        }
        // ---- A fragments (direct from global fp32, L2 resident weights) ----
        BF16x16 afr[2];
#pragma unroll
        for (int ti = 0; ti < 2; ++ti) {
            const float* ap = Ab + (size_t)(aRow + ti * 16) * K + k0 + aOff;
#pragma unroll
            for (int i = 0; i < 8; ++i) {
                afr[ti].u[i]     = f2bf(ap[i]);
                afr[ti].u[i + 8] = f2bf(ap[i + 8]);
            }
        }

        // ---- issue NEXT tile's global loads (latency hidden by WMMA) ----
        float          nf[16];
        unsigned short nu[16];
        if (hasNext) {
            const size_t ro = (size_t)b * bBatch + (size_t)(k0 + 32 + sc) * N
                            + Nbase + sp0;
            if (BF32) {
                const float* r0 = (const float*)Bsrc + ro;
                const float* r1 = r0 + N;
#pragma unroll
                for (int j = 0; j < 8; ++j) { nf[j] = r0[j]; nf[8 + j] = r1[j]; }
            } else {
                const unsigned short* r0 = (const unsigned short*)Bsrc + ro;
                const unsigned short* r1 = r0 + N;
#pragma unroll
                for (int j = 0; j < 8; ++j) { nu[j] = r0[j]; nu[8 + j] = r1[j]; }
            }
        }

        // ---- 4x v_wmma_f32_16x16x32_bf16 ----
#pragma unroll
        for (int ti = 0; ti < 2; ++ti)
#pragma unroll
            for (int tj = 0; tj < 2; ++tj)
                acc[ti][tj] = __builtin_amdgcn_wmma_f32_16x16x32_bf16(
                    false, afr[ti].v, false, bfr[tj].v,
                    (short)0, acc[ti][tj], false, false);

        // ---- commit next tile to the alternate LDS buffer ----
        if (hasNext) {
            const int nb = buf ^ 1;
#pragma unroll
            for (int j = 0; j < 8; ++j) {
                unsigned lo, hi;
                if (BF32) { lo = f2bf(nf[j]); hi = f2bf(nf[8 + j]); }
                else      { lo = nu[j];       hi = nu[8 + j]; }
                *(unsigned*)&ldsB[nb][(sp0 + j) * 32 + sc] = lo | (hi << 16);
            }
        }
        __syncthreads();
        buf ^= 1;
    }

    // ---- epilogue ----
#pragma unroll
    for (int ti = 0; ti < 2; ++ti)
#pragma unroll
        for (int tj = 0; tj < 2; ++tj) {
            const long n  = Nbase + wn * 32 + tj * 16 + (lane & 15);
            const int  r0 = Mbase + wm * 32 + ti * 16 + (lane >> 4) * 8;
#pragma unroll
            for (int r = 0; r < 8; ++r) {
                size_t idx = (size_t)b * oBatch + (size_t)(r0 + r) * N + n;
                if (OUTBF) ((unsigned short*)Out)[idx] = f2bf(acc[ti][tj][r]);
                else       ((float*)Out)[idx]          = acc[ti][tj][r];
            }
        }
}

// =====================================================================
// Depthwise 3x3 conv (SAME, groups=3C), bf16 in/out, fused sum-of-squares
// for q/k channels (needed by F.normalize) via block reduce + atomics.
// grid = (HW/256, 3C, B)
// =====================================================================
__global__ __launch_bounds__(256) void dwconv3x3_kernel(
    const unsigned short* __restrict__ in, unsigned short* __restrict__ out,
    const float* __restrict__ w, float* __restrict__ sumsq)
{
    const int ch = blockIdx.y;
    const int b  = blockIdx.z;
    const int p  = blockIdx.x * 256 + threadIdx.x;
    const int y  = p >> 8;
    const int x  = p & 255;

    const unsigned short* src = in + ((size_t)b * C3 + ch) * HWP;
    const float* wc = w + ch * 9;

    float acc = 0.f;
#pragma unroll
    for (int ky = 0; ky < 3; ++ky) {
        const int yy = y + ky - 1;
        if ((unsigned)yy < (unsigned)HH) {
#pragma unroll
            for (int kx = 0; kx < 3; ++kx) {
                const int xx = x + kx - 1;
                if ((unsigned)xx < (unsigned)WW)
                    acc += bf2f(src[yy * WW + xx]) * wc[ky * 3 + kx];
            }
        }
    }
    out[((size_t)b * C3 + ch) * HWP + p] = f2bf(acc);

    // fused ||row||^2 accumulation for q (ch<C) and k (C<=ch<2C)
    __shared__ float red[256];
    red[threadIdx.x] = (ch < 2 * CC) ? acc * acc : 0.f;
    __syncthreads();
#pragma unroll
    for (int s = 128; s > 0; s >>= 1) {
        if (threadIdx.x < s) red[threadIdx.x] += red[threadIdx.x + s];
        __syncthreads();
    }
    if (threadIdx.x == 0 && ch < 2 * CC)
        atomicAdd(&sumsq[b * 2 * CC + ch], red[0]);
}

// =====================================================================
// attn_raw[b,h,d,e] += sum_n q[d,n]*k[e,n]  — split-K WMMA.
// grid = (B*h, 16), block 256 (8 waves); each wave: K span 512, 9 tiles.
// =====================================================================
__global__ __launch_bounds__(256) void attn_qk_kernel(
    const unsigned short* __restrict__ qkv, float* __restrict__ attnRaw)
{
    const int bh   = blockIdx.x;
    const int b    = bh >> 2;
    const int head = bh & 3;
    const unsigned short* qb = qkv + ((size_t)b * C3 + head * DD) * HWP;
    const unsigned short* kb = qb + (size_t)CC * HWP;

    const int lane = threadIdx.x & 31;
    const int wave = threadIdx.x >> 5;
    const long kstart = (long)blockIdx.y * 4096 + wave * 512;

    v8f acc[3][3];
#pragma unroll
    for (int i = 0; i < 3; ++i)
#pragma unroll
        for (int j = 0; j < 3; ++j) acc[i][j] = v8f_zero();

    const int dl   = lane & 15;
    const int off8 = (lane >> 4) * 8;
    const int kh   = (lane >> 4) * 16;

    for (int it = 0; it < 16; ++it) {
        const long n0 = kstart + it * 32;
        BF16x16 a[3], bf[3];
#pragma unroll
        for (int t = 0; t < 3; ++t) {
            const unsigned short* ap = qb + (size_t)(t * 16 + dl) * HWP + n0 + kh;
            a[t].h[0] = *(const u16x8*)ap;
            a[t].h[1] = *(const u16x8*)(ap + 8);
            const unsigned short* bp = kb + (size_t)(t * 16 + dl) * HWP + n0 + kh;
            bf[t].h[0] = *(const u16x8*)bp;
            bf[t].h[1] = *(const u16x8*)(bp + 8);
        }
#pragma unroll
        for (int i = 0; i < 3; ++i)
#pragma unroll
            for (int j = 0; j < 3; ++j)
                acc[i][j] = __builtin_amdgcn_wmma_f32_16x16x32_bf16(
                    false, a[i].v, false, bf[j].v,
                    (short)0, acc[i][j], false, false);
    }

    float* ar = attnRaw + (size_t)bh * DD * DD;
#pragma unroll
    for (int i = 0; i < 3; ++i)
#pragma unroll
        for (int j = 0; j < 3; ++j) {
            const int e  = j * 16 + (lane & 15);
            const int r0 = i * 16 + (lane >> 4) * 8;
#pragma unroll
            for (int r = 0; r < 8; ++r)
                atomicAdd(&ar[(r0 + r) * DD + e], acc[i][j][r]);
        }
}

// =====================================================================
// softmax over e, with fused 1/max(||q||,eps), 1/max(||k||,eps), temp.
// grid = B*h, block 64 (one thread per row d). In place.
// =====================================================================
__global__ void softmax_kernel(float* __restrict__ attn,
                               const float* __restrict__ sumsq,
                               const float* __restrict__ temp)
{
    const int bh = blockIdx.x, b = bh >> 2, head = bh & 3;
    const int d  = threadIdx.x;
    if (d >= DD) return;
    const float* sq = sumsq + b * 2 * CC;
    const float rq = 1.f / fmaxf(sqrtf(sq[head * DD + d]), 1e-12f);
    const float tp = temp[head];
    float* row = attn + (size_t)bh * DD * DD + (size_t)d * DD;

    float vals[DD];
    float m = -3.4e38f;
#pragma unroll 1
    for (int e = 0; e < DD; ++e) {
        const float rk = 1.f / fmaxf(sqrtf(sq[CC + head * DD + e]), 1e-12f);
        vals[e] = row[e] * rq * rk * tp;
        m = fmaxf(m, vals[e]);
    }
    float s = 0.f;
#pragma unroll 1
    for (int e = 0; e < DD; ++e) { vals[e] = __expf(vals[e] - m); s += vals[e]; }
    const float inv = 1.f / s;
#pragma unroll 1
    for (int e = 0; e < DD; ++e) row[e] = vals[e] * inv;
}

// =====================================================================
// W2[b] = proj_w * blockdiag(attn_b)  (collapses attn@v and proj into one GEMM)
// W2[b][o][h*D+e] = sum_d proj_w[o][h*D+d] * attn[b,h,d,e].  grid=(B,144),256 thr
// =====================================================================
__global__ void w2_kernel(const float* __restrict__ proj_w,
                          const float* __restrict__ attn,
                          float* __restrict__ W2)
{
    const int b   = blockIdx.x;
    const int idx = blockIdx.y * 256 + threadIdx.x;   // < C*C
    const int o = idx / CC, c = idx % CC;
    const int head = c / DD, e = c % DD;
    const float* ar = attn + ((size_t)(b * NH + head)) * DD * DD;
    const float* pw = proj_w + (size_t)o * CC + head * DD;
    float s = 0.f;
#pragma unroll 1
    for (int d = 0; d < DD; ++d) s += pw[d] * ar[d * DD + e];
    W2[(size_t)b * CC * CC + idx] = s;
}

// =====================================================================
// host-side launch
// =====================================================================
extern "C" void kernel_launch(void* const* d_in, const int* in_sizes, int n_in,
                              void* d_out, int out_size, void* d_ws, size_t ws_size,
                              hipStream_t stream)
{
    const float* x      = (const float*)d_in[0];   // [B,C,H,W]
    const float* qkv_w  = (const float*)d_in[1];   // [3C,C]
    const float* dw_w   = (const float*)d_in[2];   // [3C,1,3,3]
    const float* temp   = (const float*)d_in[3];   // [h,1,1]
    const float* proj_w = (const float*)d_in[4];   // [C,C]
    float* out = (float*)d_out;                    // [B,C,H,W]

    char* ws = (char*)d_ws;
    const size_t szQKV = (size_t)BB * C3 * HWP * sizeof(unsigned short); // 302 MB
    unsigned short* qkvLin = (unsigned short*)(ws);
    unsigned short* qkvDw  = (unsigned short*)(ws + szQKV);
    float* sumsq = (float*)(ws + 2 * szQKV);                   // B*2C
    float* attnB = sumsq + (size_t)BB * 2 * CC;                // B*h*D*D (contiguous)
    float* W2    = attnB + (size_t)BB * NH * DD * DD;          // B*C*C

    // 1. zero the accumulators (sumsq + attn are contiguous: 38400 floats)
    {
        const int nz = BB * 2 * CC + BB * NH * DD * DD;
        zero_f32_kernel<<<(nz + 255) / 256, 256, 0, stream>>>(sumsq, nz);
    }

    // 2. qkv 1x1 conv: [576x192] x [192x65536] per batch -> bf16
    {
        dim3 grid(HWP / 128, C3 / 64, BB);
        gemm_wmma_kernel<true, true><<<grid, 256, 0, stream>>>(
            qkv_w, x, qkvLin, C3, CC, (long)HWP,
            /*aBatch=*/0, /*bBatch=*/(long)CC * HWP, /*oBatch=*/(long)C3 * HWP);
    }

    // 3. depthwise 3x3 + fused q/k sum-of-squares
    {
        dim3 grid(HWP / 256, C3, BB);
        dwconv3x3_kernel<<<grid, 256, 0, stream>>>(qkvLin, qkvDw, dw_w, sumsq);
    }

    // 4. attn = q.k^T, split-K WMMA with atomic f32 accumulation
    {
        dim3 grid(BB * NH, HWP / 4096);
        attn_qk_kernel<<<grid, 256, 0, stream>>>(qkvDw, attnB);
    }

    // 5. fused normalize-scale + softmax (in place)
    softmax_kernel<<<BB * NH, 64, 0, stream>>>(attnB, sumsq, temp);

    // 6. W2[b] = proj_w * blockdiag(attn_b)
    {
        dim3 grid(BB, (CC * CC) / 256);
        w2_kernel<<<grid, 256, 0, stream>>>(proj_w, attnB, W2);
    }

    // 7. y = W2[b] * v : [192x192] x [192x65536] per batch -> fp32 out
    {
        dim3 grid(HWP / 128, CC / 64, BB);
        const unsigned short* vrows = qkvDw + (size_t)2 * CC * HWP; // v at ch 2C
        gemm_wmma_kernel<false, false><<<grid, 256, 0, stream>>>(
            W2, vrows, out, CC, CC, (long)HWP,
            /*aBatch=*/(long)CC * CC, /*bBatch=*/(long)C3 * HWP,
            /*oBatch=*/(long)CC * HWP);
    }
}
